// IDRRayTracer_60404420051282
// MI455X (gfx1250) — matrix operand order repaired and checked by reading, compiled, not run on hardware
//
#include <hip/hip_runtime.h>
#include <hip/hip_bf16.h>

typedef __attribute__((ext_vector_type(16))) _Float16 v16h;
typedef __attribute__((ext_vector_type(8)))  _Float16 v8h;
typedef __attribute__((ext_vector_type(8)))  float    v8f;
typedef __attribute__((ext_vector_type(4)))  unsigned int u32x4;
typedef __attribute__((ext_vector_type(8)))  int      i32x8;
typedef __attribute__((ext_vector_type(4)))  int      i32x4;

#define HIDDEN      256
#define RAYS_PER_WG 32            // two 16-row M-tiles per workgroup
#define TRACE_ITERS 16
#define SEC_COARSE  16
#define SEC_ITERS   8
#define HPAD        (HIDDEN + 8)  // pad rows to de-phase LDS banks

// ws layout (bytes):
//   [0      , 131072) : W2 f16 B-fragment blob  (16 ntile x 8 kstep x 32 lane x 16 half)
//   [131072 , 147456) : W1 f16 B-fragment blob  (16 ntile x 32 lane x 16 half, K padded 3->32)
#define WS_W2_OFF 0
#define WS_W1_OFF 131072

// ---------------------------------------------------------------------------
// Pre-swizzle weights into per-lane WMMA B-fragment blobs (f32 -> f16).
// B-matrix 32x16 f16 k order used here (contiguous; A-frags use the same): lanes 0-15 hold K=0..15 (n=lane),
// lanes 16-31 hold K=16..31 (n=lane-16); half h within a lane is K=base+h.
// ---------------------------------------------------------------------------
__global__ void idr_prep_weights(const float* __restrict__ W1,
                                 const float* __restrict__ W2,
                                 _Float16* __restrict__ w1blob,
                                 _Float16* __restrict__ w2blob) {
  int idx = blockIdx.x * blockDim.x + threadIdx.x;
  if (idx < 16 * 8 * 32 * 16) {              // W2 blob: 65536 halves
    int h    = idx & 15;
    int lane = (idx >> 4) & 31;
    int ks   = (idx >> 9) & 7;
    int nt   = idx >> 12;
    int k = ks * 32 + ((lane >> 4) << 4) + h;
    int n = nt * 16 + (lane & 15);
    w2blob[idx] = (_Float16)W2[k * HIDDEN + n];
  }
  int j = idx - 16 * 8 * 32 * 16;
  if (j >= 0 && j < 16 * 32 * 16) {          // W1 blob: 8192 halves
    int h    = j & 15;
    int lane = (j >> 4) & 31;
    int nt   = j >> 9;
    int k = ((lane >> 4) << 4) + h;
    int n = nt * 16 + (lane & 15);
    w1blob[j] = (_Float16)((k < 3) ? W1[k * HIDDEN + n] : 0.0f);
  }
}

// ---------------------------------------------------------------------------
// Main tracer: 1 workgroup (8 waves) = 32 rays (2 M-tiles). 57 evals/ray.
// ---------------------------------------------------------------------------
__global__ __launch_bounds__(256) void idr_trace_kernel(
    const float* __restrict__ dirs, const float* __restrict__ oris,
    const int*   __restrict__ omask,
    const _Float16* __restrict__ w1blob, const _Float16* __restrict__ w2blob,
    const float* __restrict__ W3, const float* __restrict__ b1,
    const float* __restrict__ b2, const float* __restrict__ b3,
    float* __restrict__ out, int n_rays) {

  extern __shared__ __align__(16) _Float16 w2lds[];   // 65536 halves (128 KB)

  __shared__ __align__(16) _Float16 hA[RAYS_PER_WG][HPAD];  // h1 then h2 (in place)
  __shared__ float sOri[RAYS_PER_WG][3], sDir[RAYS_PER_WG][3];
  __shared__ float sZ[RAYS_PER_WG], sS[RAYS_PER_WG], sOut[RAYS_PER_WG];
  __shared__ float sB1[HIDDEN], sB2[HIDDEN];
  __shared__ __align__(16) _Float16 sW3h[HIDDEN];
  __shared__ float sB3;

  const int tid  = threadIdx.x;
  const int wave = tid >> 5;
  const int lane = tid & 31;
  const int m16  = lane & 15;
  const int base = blockIdx.x * RAYS_PER_WG;
  const int ray  = base + tid;              // valid for tid < 32

  // ---- stage biases / W3 into LDS ----
  if (tid < HIDDEN) {
    sB1[tid]  = b1[tid];
    sB2[tid]  = b2[tid];
    sW3h[tid] = (_Float16)W3[tid];
  }
  if (tid == 0) sB3 = b3[0];

  // ---- stage W2 fragment blob into LDS via Tensor Data Mover ----
#if defined(__has_builtin) && __has_builtin(__builtin_amdgcn_tensor_load_to_lds) && \
    __has_builtin(__builtin_amdgcn_s_wait_tensorcnt)
  if (wave == 0) {
    unsigned long long gaddr = (unsigned long long)(const void*)w2blob;
    unsigned ldsoff = (unsigned)(unsigned long long)(const void*)w2lds;
    // D# group0: count=1 | lds_addr | global_addr | type=2
    u32x4 g0;
    g0[0] = 1u;
    g0[1] = ldsoff;
    g0[2] = (unsigned)(gaddr & 0xffffffffu);
    g0[3] = (unsigned)((gaddr >> 32) & 0x01ffffffu) | (2u << 30);
    // D# group1: data_size=3 (8B), tensor_dim0 = tile_dim0 = 16384, stride0 = 16384
    i32x8 g1;
    g1[0] = 0x00030000;            // data_size=3, no multicast, no pad/iterate
    g1[1] = (int)(16384u << 16);   // tensor_dim0[15:0] in bits[79:48 -> w1 31:16]
    g1[2] = 0;
    g1[3] = (int)(16384u << 16);   // tile_dim0 in bits[127:112]
    g1[4] = 0;
    g1[5] = 16384;                 // tensor_dim0_stride[31:0]
    g1[6] = 0;
    g1[7] = 0;
    i32x4 gz4; gz4[0] = 0; gz4[1] = 0; gz4[2] = 0; gz4[3] = 0;
    i32x8 gz8; gz8[0] = 0; gz8[1] = 0; gz8[2] = 0; gz8[3] = 0;
    gz8[4] = 0; gz8[5] = 0; gz8[6] = 0; gz8[7] = 0;
    __builtin_amdgcn_tensor_load_to_lds(g0, g1, gz4, gz4, gz8, 0);
    __builtin_amdgcn_s_wait_tensorcnt(0);
  }
#else
  {
    const uint4* src = (const uint4*)w2blob;
    uint4*       dst = (uint4*)w2lds;
    for (int i = tid; i < 65536 / 8; i += 256) dst[i] = src[i];
  }
#endif

  // ---- ray state into LDS, per-ray trace registers (tid<32) ----
  float ox = 0.f, oy = 0.f, oz = 0.f, dx = 0.f, dy = 0.f, dz = 0.f;
  bool  objm = false, mi = false;
  float z_near = 0.f, z_far = 0.f, z_front = 0.f, z_back = 0.f;
  if (tid < RAYS_PER_WG && ray < n_rays) {
    ox = oris[ray * 3 + 0]; oy = oris[ray * 3 + 1]; oz = oris[ray * 3 + 2];
    dx = dirs[ray * 3 + 0]; dy = dirs[ray * 3 + 1]; dz = dirs[ray * 3 + 2];
    objm = (omask[ray] != 0);
    sOri[tid][0] = ox; sOri[tid][1] = oy; sOri[tid][2] = oz;
    sDir[tid][0] = dx; sDir[tid][1] = dy; sDir[tid][2] = dz;
    float od    = ox * dx + oy * dy + oz * dz;
    float under = od * od - (ox * ox + oy * oy + oz * oz - 1.0f);
    mi = under > 0.0f;
    float sq = sqrtf(fmaxf(under, 0.0f));
    z_near = fmaxf(-od - sq, 0.0f);
    z_far  = fmaxf(-od + sq, 0.0f);
    z_front = z_near;
    sZ[tid] = 0.f; sS[tid] = 0.f;
  }

  // ---- W1 B-fragments live in registers (2 N-tiles per wave) ----
  v16h w1f[2];
#pragma unroll
  for (int t = 0; t < 2; ++t)
    w1f[t] = *(const v16h*)(w1blob + (((wave * 2 + t) * 32) + lane) * 16);

  __syncthreads();

  // ---- batched SDF eval over the 32 rays (p = o + sZ*d -> sS) ----
  auto evalSDF = [&]() {
    __syncthreads();
    // ---------------- layer 1: h1 = relu(P @ W1 + b1) ----------------
    v16h a1[2];
#pragma unroll
    for (int mt = 0; mt < 2; ++mt) {
      const int ridx = mt * 16 + m16;
      float px = sOri[ridx][0] + sZ[ridx] * sDir[ridx][0];
      float py = sOri[ridx][1] + sZ[ridx] * sDir[ridx][1];
      float pz = sOri[ridx][2] + sZ[ridx] * sDir[ridx][2];
      v16h a = {};
      if (lane < 16) {
        a[0] = (_Float16)px; a[1] = (_Float16)py; a[2] = (_Float16)pz;
      }
      a1[mt] = a;
    }
#pragma unroll
    for (int t = 0; t < 2; ++t) {
      const int n  = (wave * 2 + t) * 16 + m16;
      const float bn = sB1[n];
#pragma unroll
      for (int mt = 0; mt < 2; ++mt) {
        v8f c = {};
        c = __builtin_amdgcn_wmma_f32_16x16x32_f16(false, a1[mt], false, w1f[t],
                                                   (short)0, c, false, false);
        const int mb = mt * 16 + ((lane < 16) ? 0 : 8);
#pragma unroll
        for (int r = 0; r < 8; ++r) {
          float v = c[r] + bn;
          v = v > 0.f ? v : 0.f;
          hA[mb + r][n] = (_Float16)v;
        }
      }
    }
    __syncthreads();
    // ---------------- layer 2: h2 = relu(h1 @ W2 + b2) ----------------
    // K-step outer loop: A-frags shared across N-tiles, B-frags across M-tiles.
    v8f acc[2][2] = {};   // [ntile][mtile]
#pragma unroll
    for (int ks = 0; ks < 8; ++ks) {
      const int klo = ks * 32 + ((lane >> 4) << 4);   // A-frag layout
      v16h aM[2];
#pragma unroll
      for (int mt = 0; mt < 2; ++mt) {
        v8h alo = *(const v8h*)&hA[mt * 16 + m16][klo];
        v8h ahi = *(const v8h*)&hA[mt * 16 + m16][klo + 8];
        aM[mt] = __builtin_shufflevector(alo, ahi, 0, 1, 2, 3, 4, 5, 6, 7,
                                         8, 9, 10, 11, 12, 13, 14, 15);
      }
#pragma unroll
      for (int t = 0; t < 2; ++t) {
        const v16h b =
            *(const v16h*)(w2lds + ((((wave * 2 + t) * 8) + ks) * 32 + lane) * 16);
        acc[t][0] = __builtin_amdgcn_wmma_f32_16x16x32_f16(false, aM[0], false, b,
                                                           (short)0, acc[t][0],
                                                           false, false);
        acc[t][1] = __builtin_amdgcn_wmma_f32_16x16x32_f16(false, aM[1], false, b,
                                                           (short)0, acc[t][1],
                                                           false, false);
      }
    }
    __syncthreads();   // all hA (h1) reads done -> safe to overwrite with h2
#pragma unroll
    for (int t = 0; t < 2; ++t) {
      const int n  = (wave * 2 + t) * 16 + m16;
      const float bn = sB2[n];
#pragma unroll
      for (int mt = 0; mt < 2; ++mt) {
        const int mb = mt * 16 + ((lane < 16) ? 0 : 8);
#pragma unroll
        for (int r = 0; r < 8; ++r) {
          float v = acc[t][mt][r] + bn;
          v = v > 0.f ? v : 0.f;
          hA[mb + r][n] = (_Float16)v;
        }
      }
    }
    __syncthreads();
    // ---------------- layer 3: out = h2 @ W3 + b3 ----------------
    // 256 threads = 32 rays x 8 chunks of 32 K; vectorized v8h LDS reads.
    {
      const int r = tid >> 3, c8 = tid & 7;
      const int kb = c8 * 32;
      float partial = 0.f;
#pragma unroll
      for (int q = 0; q < 4; ++q) {
        v8h hv = *(const v8h*)&hA[r][kb + q * 8];
        v8h wv = *(const v8h*)&sW3h[kb + q * 8];
#pragma unroll
        for (int k = 0; k < 8; ++k)
          partial += (float)hv[k] * (float)wv[k];
      }
      partial += __shfl_xor(partial, 1);
      partial += __shfl_xor(partial, 2);
      partial += __shfl_xor(partial, 4);
      if (c8 == 0) sOut[r] = partial + sB3;
    }
    __syncthreads();
    if (tid < RAYS_PER_WG) {
      float z  = sZ[tid];
      float qx = sOri[tid][0] + z * sDir[tid][0];
      float qy = sOri[tid][1] + z * sDir[tid][1];
      float qz = sOri[tid][2] + z * sDir[tid][2];
      float nrm = sqrtf(qx * qx + qy * qy + qz * qz);
      sS[tid] = nrm - 0.5f + 0.05f * tanhf(sOut[tid]);
    }
    __syncthreads();
  };

  // ================= sphere tracing: front =================
#pragma unroll 1
  for (int it = 0; it < TRACE_ITERS; ++it) {
    if (tid < RAYS_PER_WG) sZ[tid] = z_front;
    evalSDF();
    if (tid < RAYS_PER_WG) z_front += sS[tid];
  }
  if (tid < RAYS_PER_WG) {
    z_front = fminf(fmaxf(z_front, z_near), z_far);
    sZ[tid] = z_front;
  }
  evalSDF();
  bool unfinished = false;
  if (tid < RAYS_PER_WG) {
    unfinished = (fabsf(sS[tid]) > 1e-3f) && mi;
    z_back = z_far;
  }
  // ================= sphere tracing: back =================
#pragma unroll 1
  for (int it = 0; it < TRACE_ITERS; ++it) {
    if (tid < RAYS_PER_WG) sZ[tid] = z_back;
    evalSDF();
    if (tid < RAYS_PER_WG) z_back -= sS[tid];
  }
  bool netm = false;
  if (tid < RAYS_PER_WG) {
    z_back = fminf(fmaxf(z_back, z_near), z_far);
    netm   = z_front < z_back;
  }
  // ================= secant: coarse sign-change search =================
  float prev_z = 0.f, prev_s = 0.f, z0d = 0.f, s0d = 0.f, z1d = 0.f, s1d = 0.f;
  float zlo = 0.f, zhi = 0.f, flo = 0.f, fhi = 0.f;
  bool  found = false;
#pragma unroll 1
  for (int j = 0; j < SEC_COARSE; ++j) {
    if (tid < RAYS_PER_WG) {
      float tt = (float)j * (1.0f / (float)(SEC_COARSE - 1));
      sZ[tid] = z_front + tt * (z_back - z_front);
    }
    evalSDF();
    if (tid < RAYS_PER_WG) {
      float zc = sZ[tid], sc = sS[tid];
      if (j == 0) { z0d = zc; s0d = sc; }
      if (j == 1) { z1d = zc; s1d = sc; }
      if (j > 0 && !found && (prev_s > 0.f) && (sc <= 0.f)) {
        found = true; zlo = prev_z; zhi = zc; flo = prev_s; fhi = sc;
      }
      prev_z = zc; prev_s = sc;
    }
  }
  if (tid < RAYS_PER_WG && !found) { zlo = z0d; zhi = z1d; flo = s0d; fhi = s1d; }
  // ================= secant refinement =================
  float zmid = 0.f;
#pragma unroll 1
  for (int it = 0; it < SEC_ITERS; ++it) {
    if (tid < RAYS_PER_WG) {
      float denom = fhi - flo;
      if (fabsf(denom) < 1e-9f) denom = (denom < 0.f) ? -1e-9f : 1e-9f;
      zmid = zlo - flo * (zhi - zlo) / denom;
      sZ[tid] = zmid;
    }
    evalSDF();
    if (tid < RAYS_PER_WG) {
      float fmid = sS[tid];
      if (fmid > 0.f) { zlo = zmid; flo = fmid; }
      else            { zhi = zmid; fhi = fmid; }
    }
  }
  // ================= scatter outputs =================
  if (tid < RAYS_PER_WG && ray < n_rays) {
    bool  secm    = found && objm;
    float z_trace = unfinished ? zmid : z_front;
    bool  nm      = unfinished ? secm : netm;
    float zv = mi ? z_trace : 0.0f;
    float px = mi ? (ox + zv * dx) : 0.0f;
    float py = mi ? (oy + zv * dy) : 0.0f;
    float pz = mi ? (oz + zv * dz) : 0.0f;
    out[ray * 3 + 0] = px;
    out[ray * 3 + 1] = py;
    out[ray * 3 + 2] = pz;
    out[(size_t)3 * n_rays + ray] = zv;
    out[(size_t)4 * n_rays + ray] = (nm && mi) ? 1.0f : 0.0f;
  }
}

extern "C" void kernel_launch(void* const* d_in, const int* in_sizes, int n_in,
                              void* d_out, int out_size, void* d_ws, size_t ws_size,
                              hipStream_t stream) {
  const float* dirs = (const float*)d_in[0];
  const float* oris = (const float*)d_in[1];
  const int*   msk  = (const int*)d_in[2];
  const float* W1   = (const float*)d_in[3];
  const float* b1   = (const float*)d_in[4];
  const float* W2   = (const float*)d_in[5];
  const float* b2   = (const float*)d_in[6];
  const float* W3   = (const float*)d_in[7];
  const float* b3   = (const float*)d_in[8];
  float* out = (float*)d_out;
  const int n_rays = in_sizes[0] / 3;

  char* ws = (char*)d_ws;
  _Float16* w2blob = (_Float16*)(ws + WS_W2_OFF);
  _Float16* w1blob = (_Float16*)(ws + WS_W1_OFF);

  // 1) pre-swizzle weights into WMMA fragment blobs
  {
    int total  = 16 * 8 * 32 * 16 + 16 * 32 * 16;
    int blocks = (total + 255) / 256;
    idr_prep_weights<<<blocks, 256, 0, stream>>>(W1, W2, w1blob, w2blob);
  }
  // 2) trace: 32 rays per workgroup, W2 resident in LDS (128 KB dynamic)
  {
    size_t dyn_lds = (size_t)65536 * sizeof(_Float16);
    (void)hipFuncSetAttribute((const void*)idr_trace_kernel,
                              hipFuncAttributeMaxDynamicSharedMemorySize,
                              (int)dyn_lds);
    int blocks = (n_rays + RAYS_PER_WG - 1) / RAYS_PER_WG;
    idr_trace_kernel<<<blocks, 256, dyn_lds, stream>>>(
        dirs, oris, msk, w1blob, w2blob, W3, b1, b2, b3, out, n_rays);
  }
}
